// MolGraphEncoder_84834194031132
// MI455X (gfx1250) — matrix operand order repaired and checked by reading, not compiled, not run on hardware
//
#include <hip/hip_runtime.h>
#include <hip/hip_bf16.h>
#include <stdint.h>

// ---------------------------------------------------------------------------
// MolGraphEncoder for MI455X (gfx1250, wave32).
// GEMMs via v_wmma_f32_16x16x32_f16 (f16 in / f32 acc) with the 16-row A tile
// staged in LDS once per 256-thread block (8 waves cover 128 output columns),
// eliminating the 8x redundant global reads of the 128 MB activation buffer.
// Self-loop term (H1*dis^2) fused into the GEMM epilogue. Gather/scatter and
// softmax pooling stay f32; edge scatter uses one thread per (edge, 4 feats)
// so index loads amortize and h-reads are b128-coalesced. Atomic f32 write
// volume (~770 MB/layer into L2) is the roofline floor on this chip.
// ---------------------------------------------------------------------------

#define NNODES 500000
#define NEDGES 1000000
#define NGRAPH 16384
#define FIN    82
#define FINP   96      // FIN padded to multiple of 32 for the WMMA K-loop
#define HDIM   128
#define DDIM   256

typedef __attribute__((ext_vector_type(16))) _Float16 v16h;
typedef __attribute__((ext_vector_type(8)))  float    v8f;

union V16HQ { v16h v; uint4 q[2]; };

// ---------------- WMMA GEMM ------------------------------------------------
// C[M x Ncols] = A[M x K](f16, row-major, stride K) * Bt^T, Bt = [Ncols x K] f16.
// Grid: (M/16, Ncols/128), block 256 = 8 waves; wave w owns column tile w.
// A tile (16 x K halfs, <= 4 KB) staged in LDS, read by all 8 waves.
// If C2 != nullptr, additionally writes C2 = acc * dis[row]^2 (self-loop init).
__global__ void mg_gemm_wmma(const _Float16* __restrict__ A,
                             const _Float16* __restrict__ Bt,
                             float* __restrict__ C, int ldc,
                             const float* __restrict__ bias,
                             const float* __restrict__ dis,
                             float* __restrict__ C2, int K) {
  __shared__ __align__(16) _Float16 As[16 * 128];

  const int row0 = blockIdx.x << 4;

  // cooperative stage: 16 rows x K halfs, one uint4 (8 halfs) per thread
  const int cpr    = K >> 3;            // uint4 chunks per row (12 or 16)
  const int chunks = 16 * cpr;          // 192 or 256
  for (int c = threadIdx.x; c < chunks; c += blockDim.x) {
    int r  = c / cpr;
    int kc = (c - r * cpr) << 3;
    *(uint4*)(As + r * K + kc) = *(const uint4*)(A + (size_t)(row0 + r) * K + kc);
  }
  __syncthreads();

  const int lane = threadIdx.x & 31;
  const int wave = threadIdx.x >> 5;
  const int mn   = lane & 15;
  const int hi   = (lane >> 4) & 1;          // lanes 16..31 take k 8..15 and 24..31 of each 32-k step
  const int col0 = (blockIdx.y << 7) + (wave << 4);

  // A fragment from LDS: chunks [k0+8*hi .. +7] and [k0+16+8*hi .. +7]
  const _Float16* ap = As + mn * K + (hi ? 8 : 0);
  // B fragment: per lane (column n), chunks [k0+8*hi .. +7] and [k0+16+8*hi .. +7]
  const _Float16* bp = Bt + (size_t)(col0 + mn) * K + (hi ? 8 : 0);

  v8f acc = {};
  for (int k0 = 0; k0 < K; k0 += 32) {
    V16HQ a, b;
    a.q[0] = *(const uint4*)(ap + k0);
    a.q[1] = *(const uint4*)(ap + k0 + 16);
    b.q[0] = *(const uint4*)(bp + k0);
    b.q[1] = *(const uint4*)(bp + k0 + 16);
    acc = __builtin_amdgcn_wmma_f32_16x16x32_f16(false, a.v, false, b.v,
                                                 (short)0, acc, false, false);
  }

  const float bv = bias ? bias[col0 + mn] : 0.0f;
  const int   rbase = row0 + (hi << 3);
  float* cp = C + (size_t)rbase * ldc + col0 + mn;
#pragma unroll
  for (int r = 0; r < 8; ++r)
    cp[(size_t)r * ldc] = acc[r] + bv;

  if (C2) {                                   // fused self-loop init: H2 = H1*dis^2
    float* c2 = C2 + (size_t)rbase * ldc + col0 + mn;
#pragma unroll
    for (int r = 0; r < 8; ++r) {
      float d = dis[rbase + r];
      c2[(size_t)r * ldc] = acc[r] * d * d;
    }
  }
}

// ---------------- small elementwise / scatter kernels -----------------------

__global__ void mg_fill_deg(float* deg) {               // self-loop contributes 1
  int i = blockIdx.x * blockDim.x + threadIdx.x;
  if (i < NNODES) deg[i] = 1.0f;
}

__global__ void mg_deg_edges(const int* __restrict__ dst, float* __restrict__ deg) {
  int e = blockIdx.x * blockDim.x + threadIdx.x;
  if (e < NEDGES) atomicAdd(&deg[dst[e]], 1.0f);
}

__global__ void mg_dis(float* degdis) {                 // in place: deg -> 1/sqrt(deg)
  int i = blockIdx.x * blockDim.x + threadIdx.x;
  if (i < NNODES) degdis[i] = rsqrtf(degdis[i]);        // deg >= 1 always
}

__global__ void mg_wnorm(const int* __restrict__ src, const int* __restrict__ dst,
                         const float* __restrict__ dis, float* __restrict__ w) {
  int e = blockIdx.x * blockDim.x + threadIdx.x;
  if (e < NEDGES) w[e] = dis[src[e]] * dis[dst[e]];
}

__global__ void mg_cvt_x(const float* __restrict__ x, _Float16* __restrict__ a16) {
  int idx = blockIdx.x * blockDim.x + threadIdx.x;
  if (idx >= NNODES * FINP) return;
  int i = idx / FINP, k = idx - i * FINP;
  a16[idx] = (k < FIN) ? (_Float16)x[(size_t)i * FIN + k] : (_Float16)0.0f;
}

// W [K x Ncols] f32 -> Wt [Ncols x Kpad] f16 (transposed, zero-padded K)
__global__ void mg_cvt_w(const float* __restrict__ W, _Float16* __restrict__ Wt,
                         int K, int Kpad, int Ncols) {
  int idx = blockIdx.x * blockDim.x + threadIdx.x;
  if (idx >= Ncols * Kpad) return;
  int n = idx / Kpad, k = idx - n * Kpad;
  Wt[idx] = (k < K) ? (_Float16)W[(size_t)k * Ncols + n] : (_Float16)0.0f;
}

// edge scatter: H2[dst] += H1[src] * wnorm; one thread per (edge, 4 features)
__global__ void mg_scatter(const int* __restrict__ src, const int* __restrict__ dst,
                           const float* __restrict__ w, const float* __restrict__ h1,
                           float* __restrict__ h2) {
  size_t idx = (size_t)blockIdx.x * blockDim.x + threadIdx.x;
  if (idx >= (size_t)NEDGES * (HDIM / 4)) return;
  int e = (int)(idx >> 5), q = (int)(idx & 31) << 2;
  int s = src[e], d = dst[e];
  float wv = w[e];
  float4 hv = *(const float4*)(h1 + (size_t)s * HDIM + q);
  float* p = h2 + (size_t)d * HDIM + q;
  atomicAdd(p + 0, hv.x * wv);
  atomicAdd(p + 1, hv.y * wv);
  atomicAdd(p + 2, hv.z * wv);
  atomicAdd(p + 3, hv.w * wv);
}

// h = relu(agg + b); keep f32 in place, emit f16 copy for next GEMM (float4/b64)
__global__ void mg_biasrelu(float* __restrict__ h2, const float* __restrict__ b,
                            _Float16* __restrict__ a16) {
  size_t idx = (size_t)blockIdx.x * blockDim.x + threadIdx.x;
  if (idx >= (size_t)NNODES * (HDIM / 4)) return;
  size_t base = idx << 2;
  int    q    = (int)(idx & 31) << 2;
  float4 v  = *(const float4*)(h2 + base);
  float4 bb = *(const float4*)(b + q);
  v.x = fmaxf(v.x + bb.x, 0.0f);
  v.y = fmaxf(v.y + bb.y, 0.0f);
  v.z = fmaxf(v.z + bb.z, 0.0f);
  v.w = fmaxf(v.w + bb.w, 0.0f);
  *(float4*)(h2 + base) = v;
  union { _Float16 h[4]; uint2 u; } cv;
  cv.h[0] = (_Float16)v.x; cv.h[1] = (_Float16)v.y;
  cv.h[2] = (_Float16)v.z; cv.h[3] = (_Float16)v.w;
  *(uint2*)(a16 + base) = cv.u;
}

__global__ void mg_init_graph(unsigned* __restrict__ mkey, float* __restrict__ denom) {
  int b = blockIdx.x * blockDim.x + threadIdx.x;
  if (b < NGRAPH) { mkey[b] = 0u; denom[b] = 0.0f; }
}

__global__ void mg_zero_emb(float* __restrict__ emb) {
  size_t idx = (size_t)blockIdx.x * blockDim.x + threadIdx.x;
  if (idx < (size_t)NGRAPH * HDIM) emb[idx] = 0.0f;
}

// gate score: one wave per node; order-preserving uint key for float atomicMax
__global__ void mg_gate(const float* __restrict__ h, const float* __restrict__ gw,
                        const float* __restrict__ gb, const int* __restrict__ batch,
                        float* __restrict__ gate, unsigned* __restrict__ mkey) {
  int node = blockIdx.x * (blockDim.x >> 5) + (threadIdx.x >> 5);
  if (node >= NNODES) return;
  int lane = threadIdx.x & 31;
  float4 hv = *(const float4*)(h + (size_t)node * HDIM + lane * 4);
  float4 wv = *(const float4*)(gw + lane * 4);
  float s = hv.x * wv.x + hv.y * wv.y + hv.z * wv.z + hv.w * wv.w;
  for (int d = 16; d > 0; d >>= 1) s += __shfl_down(s, d, 32);
  if (lane == 0) {
    float g = s + gb[0];
    gate[node] = g;
    unsigned bits = __float_as_uint(g);
    unsigned key  = (bits & 0x80000000u) ? ~bits : (bits | 0x80000000u);
    atomicMax(&mkey[batch[node]], key);
  }
}

__global__ void mg_expsum(const float* __restrict__ gate, const int* __restrict__ batch,
                          const unsigned* __restrict__ mkey, float* __restrict__ ebuf,
                          float* __restrict__ denom) {
  int i = blockIdx.x * blockDim.x + threadIdx.x;
  if (i >= NNODES) return;
  int b = batch[i];
  unsigned key = mkey[b];
  unsigned bits = (key & 0x80000000u) ? (key & 0x7FFFFFFFu) : ~key;
  float m = __uint_as_float(bits);
  float ev = __expf(gate[i] - m);
  ebuf[i] = ev;
  atomicAdd(&denom[b], ev);
}

// weighted scatter into graph embeddings; one thread per (node, 4 features)
__global__ void mg_wscatter(const float* __restrict__ h, const int* __restrict__ batch,
                            const float* __restrict__ ebuf, const float* __restrict__ denom,
                            float* __restrict__ emb) {
  size_t idx = (size_t)blockIdx.x * blockDim.x + threadIdx.x;
  if (idx >= (size_t)NNODES * (HDIM / 4)) return;
  int i = (int)(idx >> 5), q = (int)(idx & 31) << 2;
  int b = batch[i];
  float alpha = ebuf[i] / denom[b];
  float4 hv = *(const float4*)(h + (size_t)i * HDIM + q);
  float* p = emb + (size_t)b * HDIM + q;
  atomicAdd(p + 0, alpha * hv.x);
  atomicAdd(p + 1, alpha * hv.y);
  atomicAdd(p + 2, alpha * hv.z);
  atomicAdd(p + 3, alpha * hv.w);
}

__global__ void mg_cvt_emb(const float* __restrict__ emb, _Float16* __restrict__ emb16) {
  size_t idx = (size_t)blockIdx.x * blockDim.x + threadIdx.x;
  if (idx < (size_t)NGRAPH * HDIM) emb16[idx] = (_Float16)emb[idx];
}

// ---------------------------------------------------------------------------

extern "C" void kernel_launch(void* const* d_in, const int* in_sizes, int n_in,
                              void* d_out, int out_size, void* d_ws, size_t ws_size,
                              hipStream_t stream) {
  (void)in_sizes; (void)n_in; (void)out_size; (void)ws_size;

  const float* x      = (const float*)d_in[0];
  const int*   ei     = (const int*)d_in[1];
  const int*   src    = ei;
  const int*   dst    = ei + NEDGES;
  const int*   batch  = (const int*)d_in[2];
  const float* W0     = (const float*)d_in[3];
  const float* b0     = (const float*)d_in[4];
  const float* W1     = (const float*)d_in[5];
  const float* b1     = (const float*)d_in[6];
  const float* W2     = (const float*)d_in[7];
  const float* b2     = (const float*)d_in[8];
  const float* gw     = (const float*)d_in[9];
  const float* gb     = (const float*)d_in[10];
  const float* projw  = (const float*)d_in[11];
  const float* projb  = (const float*)d_in[12];
  float* out = (float*)d_out;

  // -------- workspace carve-out (256B aligned) --------
  char* ws = (char*)d_ws;
  size_t off = 0;
  auto carve = [&](size_t bytes) -> void* {
    void* p = ws + off;
    off = (off + bytes + 255) & ~(size_t)255;
    return p;
  };
  _Float16* A16   = (_Float16*)carve((size_t)NNODES * HDIM * sizeof(_Float16)); // activations f16
  float*    H1    = (float*)   carve((size_t)NNODES * HDIM * sizeof(float));    // gemm out
  float*    H2    = (float*)   carve((size_t)NNODES * HDIM * sizeof(float));    // aggregated / relu
  float*    dis   = (float*)   carve((size_t)NNODES * sizeof(float));           // deg -> rsqrt(deg)
  float*    gate  = (float*)   carve((size_t)NNODES * sizeof(float));
  float*    ebuf  = (float*)   carve((size_t)NNODES * sizeof(float));
  float*    wnorm = (float*)   carve((size_t)NEDGES * sizeof(float));
  unsigned* mkey  = (unsigned*)carve((size_t)NGRAPH * sizeof(unsigned));
  float*    denom = (float*)   carve((size_t)NGRAPH * sizeof(float));
  float*    emb   = (float*)   carve((size_t)NGRAPH * HDIM * sizeof(float));
  _Float16* emb16 = (_Float16*)carve((size_t)NGRAPH * HDIM * sizeof(_Float16));
  _Float16* Wt    = (_Float16*)carve((size_t)DDIM * HDIM * sizeof(_Float16));   // max weight tile

  const int T = 256;
  auto cdiv = [](size_t a, size_t b) -> unsigned { return (unsigned)((a + b - 1) / b); };

  // -------- degrees and edge norms --------
  mg_fill_deg <<<cdiv(NNODES, T), T, 0, stream>>>(dis);
  mg_deg_edges<<<cdiv(NEDGES, T), T, 0, stream>>>(dst, dis);
  mg_dis      <<<cdiv(NNODES, T), T, 0, stream>>>(dis);
  mg_wnorm    <<<cdiv(NEDGES, T), T, 0, stream>>>(src, dst, dis, wnorm);

  // -------- layer 0 input: x(f32, 82) -> f16 padded 96 --------
  mg_cvt_x<<<cdiv((size_t)NNODES * FINP, T), T, 0, stream>>>(x, A16);

  const float* Ws[3] = { W0, W1, W2 };
  const float* bs[3] = { b0, b1, b2 };
  for (int layer = 0; layer < 3; ++layer) {
    const int K    = (layer == 0) ? FIN  : HDIM;
    const int Kpad = (layer == 0) ? FINP : HDIM;
    mg_cvt_w<<<cdiv((size_t)HDIM * Kpad, T), T, 0, stream>>>(Ws[layer], Wt, K, Kpad, HDIM);
    // GEMM + fused self-loop init (H2 = H1 * dis^2)
    mg_gemm_wmma<<<dim3(NNODES / 16, HDIM / 128), 256, 0, stream>>>(
        A16, Wt, H1, HDIM, nullptr, dis, H2, Kpad);
    mg_scatter <<<cdiv((size_t)NEDGES * (HDIM / 4), T), T, 0, stream>>>(src, dst, wnorm, H1, H2);
    mg_biasrelu<<<cdiv((size_t)NNODES * (HDIM / 4), T), T, 0, stream>>>(H2, bs[layer], A16);
  }

  // -------- attention pooling --------
  mg_init_graph<<<cdiv(NGRAPH, T), T, 0, stream>>>(mkey, denom);
  mg_zero_emb  <<<cdiv((size_t)NGRAPH * HDIM, T), T, 0, stream>>>(emb);
  mg_gate      <<<cdiv(NNODES, T / 32), T, 0, stream>>>(H2, gw, gb, batch, gate, mkey);
  mg_expsum    <<<cdiv(NNODES, T), T, 0, stream>>>(gate, batch, mkey, ebuf, denom);
  mg_wscatter  <<<cdiv((size_t)NNODES * (HDIM / 4), T), T, 0, stream>>>(H2, batch, ebuf, denom, emb);

  // -------- projection GEMM: (B x 128) @ (128 x 256) + bias --------
  mg_cvt_emb<<<cdiv((size_t)NGRAPH * HDIM, T), T, 0, stream>>>(emb, emb16);
  mg_cvt_w  <<<cdiv((size_t)DDIM * HDIM, T), T, 0, stream>>>(projw, Wt, HDIM, HDIM, DDIM);
  mg_gemm_wmma<<<dim3(NGRAPH / 16, DDIM / 128), 256, 0, stream>>>(
      emb16, Wt, out, DDIM, projb, nullptr, nullptr, HDIM);
}
